// DeepDTAF_65524021067866
// MI455X (gfx1250) — matrix operand order repaired and checked by reading, not compiled, not run on hardware
//
#include <hip/hip_runtime.h>
#include <hip/hip_bf16.h>

// ---------------------------------------------------------------------------
// DeepDTAF forward for MI455X (gfx1250, wave32, WMMA).
// Dense layers -> v_wmma_f32_16x16x32_f16, 64x64 block tile, 4 wmma/wave/step,
// LDS staged with b64 global loads + b128 ds loads. Conv blocks / GAT edge
// softmax -> VALU kernels with float atomics.
// ---------------------------------------------------------------------------

typedef __attribute__((ext_vector_type(16))) _Float16 v16h;
typedef __attribute__((ext_vector_type(8)))  _Float16 v8h;
typedef __attribute__((ext_vector_type(8)))  float    v8f;

#define BBATCH 128
#define LSEQ   1000
#define FPT    40
#define LPKT   63
#define NNODE  5120
#define NEDGE  25600
#define NGRAPH 128

// ========================= generic WMMA GEMM ===============================
// C[MxN] = act(A[MxK] @ B[KxN] + bias), A/B f32 in memory, f16 WMMA math.
// Block tile: 64(M) x 64(N), 4 waves; each wave: one 16-row strip x 4 N-subtiles.
// store_mode 0: row-major MxN.  store_mode 1: C[(b*N+n)*Lout + l], m=b*Lout+l.
// act: 0 none, 1 relu, 2 prelu(alpha[0]).
__global__ void __launch_bounds__(128)
gemm_wmma(const float* __restrict__ A, const float* __restrict__ Bw,
          const float* __restrict__ bias, const float* __restrict__ alpha,
          float* __restrict__ C, int M, int K, int N,
          int act, int store_mode, int Lout) {
    __shared__ alignas(16) _Float16 sA[64 * 32];   // [m][k]
    __shared__ alignas(16) _Float16 sBt[64 * 32];  // [n][k] (transposed)
    const int tid  = threadIdx.x;
    const int wave = tid >> 5;
    const int lane = tid & 31;
    const int tM = blockIdx.x * 64;
    const int tN = blockIdx.y * 64;
    const bool interior = (tM + 64 <= M) && (tN + 64 <= N);

    v8f acc[4] = {};

    for (int k0 = 0; k0 < K; k0 += 32) {
        __builtin_prefetch(A + (size_t)tM * K + k0 + 32, 0, 1);
        if (interior && (k0 + 32 <= K)) {
            // ---- fast branch-free staging (b64 global loads; K,N are even) ----
            // A: thread -> row r = tid/2, 16 consecutive k
            {
                int r = tid >> 1, koff = (tid & 1) * 16;
                const float2* pa = (const float2*)(A + (size_t)(tM + r) * K + k0 + koff);
                v8h h0, h1;
#pragma unroll
                for (int q = 0; q < 4; ++q) {
                    float2 v = pa[q];
                    h0[2 * q] = (_Float16)v.x; h0[2 * q + 1] = (_Float16)v.y;
                }
#pragma unroll
                for (int q = 0; q < 4; ++q) {
                    float2 v = pa[4 + q];
                    h1[2 * q] = (_Float16)v.x; h1[2 * q + 1] = (_Float16)v.y;
                }
                v8h* dp = (v8h*)(sA + r * 32 + koff);
                dp[0] = h0; dp[1] = h1;
            }
            // B: thread -> one k = tid/4, 16 consecutive n; store transposed
            {
                int kk = tid >> 2, nn0 = (tid & 3) * 16;
                const float2* pb = (const float2*)(Bw + (size_t)(k0 + kk) * N + tN + nn0);
#pragma unroll
                for (int q = 0; q < 8; ++q) {
                    float2 v = pb[q];
                    sBt[(nn0 + 2 * q) * 32 + kk]     = (_Float16)v.x;
                    sBt[(nn0 + 2 * q + 1) * 32 + kk] = (_Float16)v.y;
                }
            }
        } else {
            // ---- edge tiles: scalar bounds-checked staging ----
            for (int i = tid; i < 64 * 32; i += 128) {
                int r = i >> 5, kk = i & 31;
                int gm = tM + r, gk = k0 + kk;
                float v = (gm < M && gk < K) ? A[(size_t)gm * K + gk] : 0.0f;
                sA[i] = (_Float16)v;
            }
            for (int i = tid; i < 32 * 64; i += 128) {
                int kk = i >> 6, nn = i & 63;
                int gk = k0 + kk, gn = tN + nn;
                float v = (gk < K && gn < N) ? Bw[(size_t)gk * N + gn] : 0.0f;
                sBt[nn * 32 + kk] = (_Float16)v;
            }
        }
        __syncthreads();

        const int hh   = lane >> 4;
        const int mrow = (wave << 4) + (lane & 15);
        // A fragment: K = 16*hh + 0..15 (contiguous, matches B)
        const v8h* pa = (const v8h*)(sA + mrow * 32 + 16 * hh);
        v16h a = __builtin_shufflevector(pa[0], pa[1],
                 0, 1, 2, 3, 4, 5, 6, 7, 8, 9, 10, 11, 12, 13, 14, 15);
#pragma unroll
        for (int t = 0; t < 4; ++t) {
            // B fragment: column n, K = 16*hh + 0..15 (contiguous in sBt)
            const v8h* pb = (const v8h*)(sBt + (t * 16 + (lane & 15)) * 32 + 16 * hh);
            v16h b = __builtin_shufflevector(pb[0], pb[1],
                     0, 1, 2, 3, 4, 5, 6, 7, 8, 9, 10, 11, 12, 13, 14, 15);
            acc[t] = __builtin_amdgcn_wmma_f32_16x16x32_f16(false, a, false, b,
                                                            (short)0, acc[t], false, false);
        }
        __syncthreads();
    }

    // epilogue: C/D layout: lane -> N=lane%16, VGPR r -> M = r + 8*(lane/16)
    const int hh = lane >> 4;
#pragma unroll
    for (int t = 0; t < 4; ++t) {
        int n = tN + t * 16 + (lane & 15);
#pragma unroll
        for (int r = 0; r < 8; ++r) {
            int m = tM + (wave << 4) + r + 8 * hh;
            if (m < M && n < N) {
                float v = acc[t][r];
                if (bias) v += bias[n];
                if (act == 1)      v = v > 0.0f ? v : 0.0f;
                else if (act == 2) { float al = alpha[0]; v = v > 0.0f ? v : al * v; }
                if (store_mode == 0) {
                    C[(size_t)m * N + n] = v;
                } else {
                    int bq = m / Lout, l = m % Lout;
                    C[((size_t)bq * N + n) * Lout + l] = v;
                }
            }
        }
    }
}

// ========================= conv-block kernels ==============================
__global__ void c1_bn_prelu(const float* __restrict__ x, const float* __restrict__ w,
                            const float* __restrict__ bias, const float* __restrict__ g,
                            const float* __restrict__ bb, const float* __restrict__ alpha,
                            float* __restrict__ o, int Cin, int n, int L) {
    int idx = blockIdx.x * blockDim.x + threadIdx.x;
    if (idx >= BBATCH * L) return;
    int b = idx / L, l = idx % L;
    float al = alpha[0];
    float scale = rsqrtf(1.0f + 1e-5f);
    const float* xb = x + (size_t)b * Cin * L + l;
    for (int j = 0; j < n; ++j) {
        float s = bias[j];
        const float* wr = w + (size_t)j * Cin;
        for (int ci = 0; ci < Cin; ++ci) s += wr[ci] * xb[(size_t)ci * L];
        float y = s * (g[j] * scale) + bb[j];
        o[((size_t)b * n + j) * L + l] = y > 0.0f ? y : al * y;
    }
}

__global__ void dil_combo(const float* __restrict__ o, const float* __restrict__ xin,
                          const float* __restrict__ d1w, const float* __restrict__ d2w,
                          const float* __restrict__ d4w, const float* __restrict__ d8w,
                          const float* __restrict__ d16w,
                          const float* __restrict__ g2, const float* __restrict__ b2,
                          const float* __restrict__ alpha,
                          float* __restrict__ out, int n, int n1, int L) {
    int idx = blockIdx.x * blockDim.x + threadIdx.x;
    if (idx >= BBATCH * L) return;
    int b = idx / L, l = idx % L;
    int nOut = n1 + 4 * n;
    float al = alpha[0];
    float scale = rsqrtf(1.0f + 1e-5f);

    auto cv = [&](const float* W, int c, int dil) -> float {
        float s = 0.0f;
        const float* ob = o + (size_t)b * n * L;
        const float* wr = W + (size_t)c * n * 3;
        int lm = l - dil, lp = l + dil;
        for (int ci = 0; ci < n; ++ci) {
            const float* orow = ob + (size_t)ci * L;
            if (lm >= 0) s += wr[ci * 3 + 0] * orow[lm];
            s += wr[ci * 3 + 1] * orow[l];
            if (lp < L)  s += wr[ci * 3 + 2] * orow[lp];
        }
        return s;
    };
    auto emit = [&](int ch, float val) {
        if (xin) val += xin[((size_t)b * nOut + ch) * L + l];
        float y = val * (g2[ch] * scale) + b2[ch];
        out[((size_t)b * nOut + ch) * L + l] = y > 0.0f ? y : al * y;
    };

    for (int c = 0; c < n1; ++c) emit(c, cv(d1w, c, 1));
    for (int c = 0; c < n; ++c) {
        float v2 = cv(d2w, c, 2), v4 = cv(d4w, c, 4);
        float v8 = cv(d8w, c, 8), v16 = cv(d16w, c, 16);
        float a2v = v2 + v4, a3v = a2v + v8, a4v = a3v + v16;
        emit(n1 + c, v2);
        emit(n1 + n + c, a2v);
        emit(n1 + 2 * n + c, a3v);
        emit(n1 + 3 * n + c, a4v);
    }
}

__global__ void pkt_conv(const float* __restrict__ in, const float* __restrict__ w,
                         const float* __restrict__ bias, const float* __restrict__ g,
                         const float* __restrict__ be, const float* __restrict__ alpha,
                         float* __restrict__ out, int Cin, int Cout, int Lin) {
    int Lout = Lin - 2;
    int idx = blockIdx.x * blockDim.x + threadIdx.x;
    if (idx >= BBATCH * Cout * Lout) return;
    int b = idx / (Cout * Lout);
    int r = idx % (Cout * Lout);
    int oc = r / Lout, l = r % Lout;
    float s = bias[oc];
    const float* ib = in + (size_t)b * Cin * Lin;
    const float* wr = w + (size_t)oc * Cin * 3;
    for (int ci = 0; ci < Cin; ++ci) {
        const float* irow = ib + (size_t)ci * Lin + l;
        s += wr[ci * 3 + 0] * irow[0] + wr[ci * 3 + 1] * irow[1] + wr[ci * 3 + 2] * irow[2];
    }
    float scale = rsqrtf(1.0f + 1e-5f);
    float y = s * (g[oc] * scale) + be[oc];
    float al = alpha[0];
    out[((size_t)b * Cout + oc) * Lout + l] = y > 0.0f ? y : al * y;
}

__global__ void maxpool_L(const float* __restrict__ in, float* __restrict__ out,
                          int BC, int L) {
    int idx = blockIdx.x * blockDim.x + threadIdx.x;
    if (idx >= BC) return;
    const float* row = in + (size_t)idx * L;
    float m = -3.4e38f;
    for (int l = 0; l < L; ++l) m = fmaxf(m, row[l]);
    out[idx] = m;
}

// ============================ GAT kernels ==================================
__device__ inline unsigned fenc(float x) {
    unsigned u = __float_as_uint(x);
    return (u & 0x80000000u) ? ~u : (u | 0x80000000u);
}
__device__ inline float fdec(unsigned u) {
    unsigned v = (u & 0x80000000u) ? (u & 0x7FFFFFFFu) : ~u;
    return __uint_as_float(v);
}

__global__ void fill_f(float* p, int n, float v) {
    int i = blockIdx.x * blockDim.x + threadIdx.x;
    if (i < n) p[i] = v;
}
__global__ void fill_u(unsigned* p, int n, unsigned v) {
    int i = blockIdx.x * blockDim.x + threadIdx.x;
    if (i < n) p[i] = v;
}

__global__ void gat_scores(const float* __restrict__ hl, const float* __restrict__ as,
                           const float* __restrict__ ad, float* __restrict__ asrc,
                           float* __restrict__ adst, int H, int oc) {
    int idx = blockIdx.x * blockDim.x + threadIdx.x;
    if (idx >= NNODE * H) return;
    int n = idx / H, h = idx % H;
    const float* hr = hl + (size_t)n * H * oc + (size_t)h * oc;
    float s0 = 0.0f, s1 = 0.0f;
    for (int c = 0; c < oc; ++c) {
        s0 += hr[c] * as[h * oc + c];
        s1 += hr[c] * ad[h * oc + c];
    }
    asrc[idx] = s0;
    adst[idx] = s1;
}

__device__ inline void edge_sd(const int* ei, int e, int& s, int& d) {
    if (e < NEDGE) { s = ei[e]; d = ei[NEDGE + e]; }
    else           { s = e - NEDGE; d = e - NEDGE; }   // self loops
}

__global__ void gat_edge_max(const int* __restrict__ ei, const float* __restrict__ asrc,
                             const float* __restrict__ adst, unsigned* __restrict__ m, int H) {
    int idx = blockIdx.x * blockDim.x + threadIdx.x;
    if (idx >= (NEDGE + NNODE) * H) return;
    int e = idx / H, h = idx % H;
    int s, d; edge_sd(ei, e, s, d);
    float v = asrc[s * H + h] + adst[d * H + h];
    v = v > 0.0f ? v : 0.2f * v;
    atomicMax(&m[d * H + h], fenc(v));
}

__global__ void gat_edge_sum(const int* __restrict__ ei, const float* __restrict__ asrc,
                             const float* __restrict__ adst, const unsigned* __restrict__ m,
                             float* __restrict__ z, int H) {
    int idx = blockIdx.x * blockDim.x + threadIdx.x;
    if (idx >= (NEDGE + NNODE) * H) return;
    int e = idx / H, h = idx % H;
    int s, d; edge_sd(ei, e, s, d);
    float v = asrc[s * H + h] + adst[d * H + h];
    v = v > 0.0f ? v : 0.2f * v;
    atomicAdd(&z[d * H + h], __expf(v - fdec(m[d * H + h])));
}

__global__ void gat_edge_accum(const int* __restrict__ ei, const float* __restrict__ asrc,
                               const float* __restrict__ adst, const unsigned* __restrict__ m,
                               const float* __restrict__ z, const float* __restrict__ hl,
                               float* __restrict__ aggr, int H, int oc) {
    int idx = blockIdx.x * blockDim.x + threadIdx.x;
    if (idx >= (NEDGE + NNODE) * H) return;
    int e = idx / H, h = idx % H;
    int s, d; edge_sd(ei, e, s, d);
    float v = asrc[s * H + h] + adst[d * H + h];
    v = v > 0.0f ? v : 0.2f * v;
    float alpha = __expf(v - fdec(m[d * H + h])) / (z[d * H + h] + 1e-16f);
    const float* hr = hl + (size_t)s * H * oc + (size_t)h * oc;
    float* ar = aggr + (size_t)d * H * oc + (size_t)h * oc;
    for (int c = 0; c < oc; ++c) atomicAdd(&ar[c], alpha * hr[c]);
}

__global__ void gat_finalize(const float* __restrict__ aggr, const float* __restrict__ bias,
                             float* __restrict__ hout, int F, int act) {
    int idx = blockIdx.x * blockDim.x + threadIdx.x;
    if (idx >= NNODE * F) return;
    int f = idx % F;
    float v = aggr[idx] + bias[f];
    if (act == 3)      v = v > 0.0f ? v : expm1f(v);
    else if (act == 1) v = v > 0.0f ? v : 0.0f;
    hout[idx] = v;
}

__global__ void graph_pool(const float* __restrict__ h, const int* __restrict__ batch,
                           unsigned* __restrict__ gp, int F) {
    int idx = blockIdx.x * blockDim.x + threadIdx.x;
    if (idx >= NNODE * F) return;
    int n = idx / F, f = idx % F;
    atomicMax(&gp[(size_t)batch[n] * F + f], __float_as_uint(h[idx]));
}

__global__ void u2f(const unsigned* __restrict__ in, float* __restrict__ out, int n) {
    int i = blockIdx.x * blockDim.x + threadIdx.x;
    if (i < n) out[i] = __uint_as_float(in[i]);
}

__global__ void concat3(const float* __restrict__ a, const float* __restrict__ b,
                        const float* __restrict__ c, float* __restrict__ out) {
    int idx = blockIdx.x * blockDim.x + threadIdx.x;
    if (idx >= BBATCH * 384) return;
    int bq = idx / 384, f = idx % 384;
    float v;
    if (f < 128)      v = a[bq * 128 + f];
    else if (f < 256) v = b[bq * 128 + (f - 128)];
    else              v = c[bq * 128 + (f - 256)];
    out[idx] = v;
}

// =============================== launch ====================================
static inline int gdiv(int a, int b) { return (a + b - 1) / b; }

extern "C" void kernel_launch(void* const* d_in, const int* in_sizes, int n_in,
                              void* d_out, int out_size, void* d_ws, size_t ws_size,
                              hipStream_t stream) {
    (void)in_sizes; (void)n_in; (void)ws_size; (void)out_size;
    // ---- unpack inputs (setup_inputs() recursive insertion order) ----
    int idx = 0;
    auto F = [&]() { return (const float*)d_in[idx++]; };
    const float* seq = F();
    const float* pkt = F();
    const float* x   = F();
    const float* se_w = F(); const float* se_b = F();
    const float* pe_w = F(); const float* pe_b = F();
    const float* cs[4][13];
    for (int b = 0; b < 4; ++b) for (int j = 0; j < 13; ++j) cs[b][j] = F();
    const float* cp[3][5];
    for (int b = 0; b < 3; ++b) for (int j = 0; j < 5; ++j) cp[b][j] = F();
    const float* gt[5][4];
    for (int b = 0; b < 5; ++b) for (int j = 0; j < 4; ++j) gt[b][j] = F();
    const float* fc_w = F(); const float* fc_b = F();
    const float* cl[3][3];
    for (int b = 0; b < 3; ++b) for (int j = 0; j < 3; ++j) cl[b][j] = F();
    const int* ei    = (const int*)d_in[idx++];
    const int* batch = (const int*)d_in[idx++];

    // ---- workspace arena ----
    char* ws = (char*)d_ws;
    size_t off = 0;
    auto alloc = [&](size_t nfl) -> float* {
        float* p = (float*)(ws + off);
        off += ((nfl * 4 + 255) / 256) * 256;
        return p;
    };
    float* A  = alloc(16384000);   // 128*128*1000 activation ping
    float* Bb = alloc(16384000);   // pong
    float* Cc = alloc(3276800);    // block-internal 'o' (<= 128*25*1000)
    float* seqc = alloc(16384);
    float* pktc = alloc(16384);
    float* gbuf = alloc(16384);
    float* catb = alloc(BBATCH * 384);
    float* t1   = alloc(16384);
    float* t2   = alloc(8192);
    // pkt + graph buffers alias region A (used only after seq branch pooled)
    float*    pe0  = A;
    float*    pe1  = A + 1048576;
    float*    h0   = A + 2097152;
    float*    h1   = A + 3932160;
    float*    hlb  = A + 5767168;
    float*    aggr = A + 7602176;
    float*    asrc = A + 9437184;
    float*    adst = asrc + 32768;
    unsigned* mb   = (unsigned*)(adst + 32768);
    float*    zb   = adst + 65536;
    unsigned* gp   = (unsigned*)(zb + 32768);

    auto gemm = [&](const float* Ag, const float* Bg, const float* bias, const float* alpha,
                    float* Cg, int M, int K, int N, int act, int smode, int Lq) {
        dim3 grid((unsigned)gdiv(M, 64), (unsigned)gdiv(N, 64));
        gemm_wmma<<<grid, 128, 0, stream>>>(Ag, Bg, bias, alpha, Cg, M, K, N, act, smode, Lq);
    };

    // ================= protein sequence branch =================
    gemm(seq, se_w, se_b, nullptr, A, BBATCH * LSEQ, FPT, 128, 0, 1, LSEQ);
    const int CinArr[4] = {128, 32, 64, 64};
    const int nArr[4]   = {6, 12, 12, 25};
    const int n1Arr[4]  = {8, 16, 16, 28};
    const bool addArr[4] = {false, false, true, false};
    float* cur = A; float* nxt = Bb;
    const int gBL = gdiv(BBATCH * LSEQ, 256);
    for (int b = 0; b < 4; ++b) {
        c1_bn_prelu<<<gBL, 256, 0, stream>>>(cur, cs[b][0], cs[b][1], cs[b][2], cs[b][3],
                                             cs[b][4], Cc, CinArr[b], nArr[b], LSEQ);
        dil_combo<<<gBL, 256, 0, stream>>>(Cc, addArr[b] ? cur : nullptr,
                                           cs[b][5], cs[b][6], cs[b][7], cs[b][8], cs[b][9],
                                           cs[b][10], cs[b][11], cs[b][12],
                                           nxt, nArr[b], n1Arr[b], LSEQ);
        float* t = cur; cur = nxt; nxt = t;
    }
    maxpool_L<<<gdiv(BBATCH * 128, 256), 256, 0, stream>>>(cur, seqc, BBATCH * 128, LSEQ);

    // ================= pocket branch =================
    gemm(pkt, pe_w, pe_b, nullptr, pe0, BBATCH * LPKT, FPT, 128, 0, 1, LPKT);
    pkt_conv<<<gdiv(BBATCH * 32 * 61, 256), 256, 0, stream>>>(pe0, cp[0][0], cp[0][1],
        cp[0][2], cp[0][3], cp[0][4], pe1, 128, 32, 63);
    pkt_conv<<<gdiv(BBATCH * 64 * 59, 256), 256, 0, stream>>>(pe1, cp[1][0], cp[1][1],
        cp[1][2], cp[1][3], cp[1][4], pe0, 32, 64, 61);
    pkt_conv<<<gdiv(BBATCH * 128 * 57, 256), 256, 0, stream>>>(pe0, cp[2][0], cp[2][1],
        cp[2][2], cp[2][3], cp[2][4], pe1, 64, 128, 59);
    maxpool_L<<<gdiv(BBATCH * 128, 256), 256, 0, stream>>>(pe1, pktc, BBATCH * 128, 57);

    // ================= ligand graph branch (5 GAT layers) =================
    const int inD[5] = {78, 312, 312, 312, 312};
    const int Hh[5]  = {4, 4, 4, 4, 1};
    const int ocA[5] = {78, 78, 78, 78, 128};
    const float* hcur = x;
    float* hping = h0; float* hpong = h1;
    const int ETOT = NEDGE + NNODE;
    for (int i = 0; i < 5; ++i) {
        int H = Hh[i], oc = ocA[i], Fo = H * oc;
        gemm(hcur, gt[i][0], nullptr, nullptr, hlb, NNODE, inD[i], Fo, 0, 0, 0);
        gat_scores<<<gdiv(NNODE * H, 256), 256, 0, stream>>>(hlb, gt[i][1], gt[i][2],
                                                             asrc, adst, H, oc);
        fill_u<<<gdiv(NNODE * H, 256), 256, 0, stream>>>(mb, NNODE * H, 0u);
        fill_f<<<gdiv(NNODE * H, 256), 256, 0, stream>>>(zb, NNODE * H, 0.0f);
        fill_f<<<gdiv(NNODE * Fo, 256), 256, 0, stream>>>(aggr, NNODE * Fo, 0.0f);
        gat_edge_max<<<gdiv(ETOT * H, 256), 256, 0, stream>>>(ei, asrc, adst, mb, H);
        gat_edge_sum<<<gdiv(ETOT * H, 256), 256, 0, stream>>>(ei, asrc, adst, mb, zb, H);
        gat_edge_accum<<<gdiv(ETOT * H, 256), 256, 0, stream>>>(ei, asrc, adst, mb, zb,
                                                                hlb, aggr, H, oc);
        gat_finalize<<<gdiv(NNODE * Fo, 256), 256, 0, stream>>>(aggr, gt[i][3], hping, Fo,
                                                                (i < 4) ? 3 : 1);
        hcur = hping;
        float* t = hping; hping = hpong; hpong = t;
    }
    fill_u<<<gdiv(NGRAPH * 128, 256), 256, 0, stream>>>(gp, NGRAPH * 128, 0u);
    graph_pool<<<gdiv(NNODE * 128, 256), 256, 0, stream>>>(hcur, batch, gp, 128);
    float* poolf = hping;
    u2f<<<gdiv(NGRAPH * 128, 256), 256, 0, stream>>>(gp, poolf, NGRAPH * 128);
    gemm(poolf, fc_w, fc_b, nullptr, gbuf, NGRAPH, 128, 128, 1 /*relu*/, 0, 0);

    // ================= fusion + classifier =================
    concat3<<<gdiv(BBATCH * 384, 256), 256, 0, stream>>>(seqc, pktc, gbuf, catb);
    gemm(catb, cl[0][0], cl[0][1], cl[0][2], t1, BBATCH, 384, 128, 2, 0, 0);
    gemm(t1,   cl[1][0], cl[1][1], cl[1][2], t2, BBATCH, 128, 64, 2, 0, 0);
    gemm(t2,   cl[2][0], cl[2][1], cl[2][2], (float*)d_out, BBATCH, 64, 1, 2, 0, 0);
}
